// GNNStream_24086176596260
// MI455X (gfx1250) — compile-verified, matrix operand order checked by reading, not run on hardware
//
#include <hip/hip_runtime.h>
#include <hip/hip_bf16.h>
#include <stdint.h>

// ---------------------------------------------------------------------------
// Problem constants (from reference)
// ---------------------------------------------------------------------------
#define BATCH 8
#define NPTS  2048
#define CLIPD 512
#define DDIM  256
#define BDIM  128
#define NCLS  10
#define MROWS (BATCH * NPTS)   // 16384 flattened rows

typedef __attribute__((ext_vector_type(16))) _Float16 v16h;
typedef __attribute__((ext_vector_type(8)))  float    v8f;

// gfx1250 async global->LDS DMA (ASYNCcnt path). Guarded so the build never
// breaks if the toolchain lacks the builtins. Parameter types learned from
// compiler diagnostics:
//   arg0: generic pointer to 16B int vector   (global source)
//   arg1: addrspace(3) pointer to 16B int vec (LDS destination)
//   arg2: imm offset, arg3: cpol
#if defined(__gfx1250__) &&                                             \
    __has_builtin(__builtin_amdgcn_global_load_async_to_lds_b128) &&    \
    __has_builtin(__builtin_amdgcn_s_wait_asynccnt)
#define ASYNC_LDS 1
typedef int v4i_vs __attribute__((vector_size(16)));
typedef __attribute__((address_space(3))) v4i_vs lds_v4i_t;
#else
#define ASYNC_LDS 0
#endif

__device__ __forceinline__ int lane_id() { return (int)(threadIdx.x & 31); }

// WMMA wrapper: D = A(16x32 f16) * B(32x16 f16) + C(f32)
__device__ __forceinline__ v8f wmma_f16(v16h a, v16h b, v8f c) {
  return __builtin_amdgcn_wmma_f32_16x16x32_f16(
      /*neg_a=*/false, a, /*neg_b=*/false, b,
      /*c_mod=*/(short)0, c, /*reuse_a=*/false, /*reuse_b=*/false);
}

// K index for element e of a v16h fragment (per CDNA5 ISA 16-bit A layout)
__device__ __forceinline__ int frag_k(int e, int lane) {
  return (e & 7) + ((e >> 3) << 4) + ((lane >> 4) << 3);
}

// A fragment: 16(M) x 32(K), row-major f16 source, leading dim ld (halves)
__device__ __forceinline__ v16h frag_a_f16(const _Float16* p, int ld) {
  const int l = lane_id();
  const int row = l & 15;
  v16h a;
#pragma unroll
  for (int e = 0; e < 16; ++e) a[e] = p[row * ld + frag_k(e, l)];
  return a;
}

// A fragment from an f32 row-major source (convert on load)
__device__ __forceinline__ v16h frag_a_f32(const float* p, int ld) {
  const int l = lane_id();
  const int row = l & 15;
  v16h a;
#pragma unroll
  for (int e = 0; e < 16; ++e) a[e] = (_Float16)p[row * ld + frag_k(e, l)];
  return a;
}

// B fragment: 32(K) x 16(N), source stored row-major K x N with ld halves/row
__device__ __forceinline__ v16h frag_b_kn(const _Float16* p, int ld) {
  const int l = lane_id();
  const int col = l & 15;
  v16h b;
#pragma unroll
  for (int e = 0; e < 16; ++e) b[e] = p[frag_k(e, l) * ld + col];
  return b;
}

// B fragment for transposed use: source stored row-major N x K (e.g. keys
// [key][d]) so that B[k][n] = p[n*ld + k]  (computes A * srcT)
__device__ __forceinline__ v16h frag_b_nk(const _Float16* p, int ld) {
  const int l = lane_id();
  const int col = l & 15;
  v16h b;
#pragma unroll
  for (int e = 0; e < 16; ++e) b[e] = p[col * ld + frag_k(e, l)];
  return b;
}

// Store a C/D fragment (16x16 f32) to row-major memory with leading dim ld
__device__ __forceinline__ void frag_c_store(float* p, int ld, v8f c) {
  const int l = lane_id();
  const int n  = l & 15;
  const int mb = (l >> 4) << 3;
#pragma unroll
  for (int r = 0; r < 8; ++r) p[(mb + r) * ld + n] = c[r];
}

// ---------------------------------------------------------------------------
// K0: zero the pooled-sum accumulator (graph-safe re-init each launch)
// ---------------------------------------------------------------------------
__global__ void k_zero(float* __restrict__ p, int n) {
  int i = blockIdx.x * blockDim.x + threadIdx.x;
  if (i < n) p[i] = 0.0f;
}

// ---------------------------------------------------------------------------
// K1: h = relu(LN(local_feat @ enc_w + enc_b))  -> f16
//     block = 16 rows x 256 cols, 8 waves, each wave owns two 16x16 tiles
// ---------------------------------------------------------------------------
__global__ __launch_bounds__(256) void k_encoder(
    const float* __restrict__ lf, const float* __restrict__ enc_w,
    const float* __restrict__ enc_b, const float* __restrict__ enc_g,
    const float* __restrict__ enc_beta, _Float16* __restrict__ h_out) {
  __shared__ _Float16 Bs[32 * 264];
  __shared__ float Cs[16 * 264];
  __shared__ float redS[256], redS2[256];
  __shared__ float rowMu[16], rowRs[16];

  const int tid  = threadIdx.x;
  const int wave = tid >> 5;
  const int mb   = blockIdx.x * 16;
  const float* Arow = lf + (size_t)mb * CLIPD;

  v8f acc0 = {}, acc1 = {};
  const int nb0 = wave * 32;
  const int nb1 = nb0 + 16;

  for (int kk = 0; kk < CLIPD; kk += 32) {
    for (int r = 0; r < 32; ++r)
      Bs[r * 264 + tid] = (_Float16)enc_w[(size_t)(kk + r) * DDIM + tid];
    __syncthreads();
    v16h a  = frag_a_f32(Arow + kk, CLIPD);
    acc0 = wmma_f16(a, frag_b_kn(Bs + nb0, 264), acc0);
    acc1 = wmma_f16(a, frag_b_kn(Bs + nb1, 264), acc1);
    __syncthreads();
  }
  {
    const int n = tid & 15;
    const float b0 = enc_b[nb0 + n], b1 = enc_b[nb1 + n];
#pragma unroll
    for (int r = 0; r < 8; ++r) { acc0[r] += b0; acc1[r] += b1; }
  }
  frag_c_store(Cs + nb0, 264, acc0);
  frag_c_store(Cs + nb1, 264, acc1);
  __syncthreads();

  // LayerNorm over 256 per row: 16 threads/row, 16 cols each
  const int r  = tid >> 4;
  const int c0 = (tid & 15) * 16;
  float s = 0.f, s2 = 0.f;
#pragma unroll
  for (int j = 0; j < 16; ++j) {
    float v = Cs[r * 264 + c0 + j];
    s += v; s2 += v * v;
  }
  redS[tid] = s; redS2[tid] = s2;
  __syncthreads();
  if ((tid & 15) == 0) {
    float S = 0.f, S2 = 0.f;
    for (int j = 0; j < 16; ++j) { S += redS[(r << 4) + j]; S2 += redS2[(r << 4) + j]; }
    float mu  = S / (float)DDIM;
    float var = fmaxf(S2 / (float)DDIM - mu * mu, 0.0f);
    rowMu[r] = mu;
    rowRs[r] = rsqrtf(var + 1e-5f);
  }
  __syncthreads();
#pragma unroll
  for (int j = 0; j < 16; ++j) {
    int cc = c0 + j;
    float v = (Cs[r * 264 + cc] - rowMu[r]) * rowRs[r] * enc_g[cc] + enc_beta[cc];
    h_out[(size_t)(mb + r) * DDIM + cc] = (_Float16)fmaxf(v, 0.0f);
  }
}

// ---------------------------------------------------------------------------
// K2: y = h @ W + b  (D=256 -> D=256), output f16. Used for q, k, v.
// ---------------------------------------------------------------------------
__global__ __launch_bounds__(256) void k_linear_qkv(
    const _Float16* __restrict__ h, const float* __restrict__ w,
    const float* __restrict__ bias, _Float16* __restrict__ out) {
  __shared__ _Float16 Bs[32 * 264];
  const int tid  = threadIdx.x;
  const int wave = tid >> 5;
  const int mb   = blockIdx.x * 16;

  v8f acc0 = {}, acc1 = {};
  const int nb0 = wave * 32;
  const int nb1 = nb0 + 16;

  for (int kk = 0; kk < DDIM; kk += 32) {
    for (int r = 0; r < 32; ++r)
      Bs[r * 264 + tid] = (_Float16)w[(size_t)(kk + r) * DDIM + tid];
    __syncthreads();
    v16h a = frag_a_f16(h + (size_t)mb * DDIM + kk, DDIM);
    acc0 = wmma_f16(a, frag_b_kn(Bs + nb0, 264), acc0);
    acc1 = wmma_f16(a, frag_b_kn(Bs + nb1, 264), acc1);
    __syncthreads();
  }
  const int l = lane_id();
  const int n = l & 15;
  const int mbase = (l >> 4) << 3;
  const float b0 = bias[nb0 + n], b1 = bias[nb1 + n];
#pragma unroll
  for (int r = 0; r < 8; ++r) {
    size_t gm = (size_t)(mb + mbase + r);
    out[gm * DDIM + nb0 + n] = (_Float16)(acc0[r] + b0);
    out[gm * DDIM + nb1 + n] = (_Float16)(acc1[r] + b1);
  }
}

// ---------------------------------------------------------------------------
// K3: streaming attention with spatial adjacency + key-mask (flash style).
//   scores = (q kT)/16; online softmax (Z over ALL keys, per reference);
//   p = exp(s-m) * exp(-dist*|gamma|) * !mask[key];  agg = (sum p v)/Z
// One block = 16 query rows; 64 chunks of 32 keys; waves 0-1 compute scores,
// all 8 waves own two 16x16 columns of the 16x256 agg accumulator.
// K/V chunks are staged into LDS with GLOBAL_LOAD_ASYNC_TO_LDS_B128 when the
// toolchain exposes it (ASYNCcnt-tracked DMA overlapping the WMMA stream).
// ---------------------------------------------------------------------------
__global__ __launch_bounds__(256) void k_attn(
    const _Float16* __restrict__ q, const _Float16* __restrict__ kmat,
    const _Float16* __restrict__ vmat, const float* __restrict__ coords,
    const int* __restrict__ mask, const float* __restrict__ gamma_p,
    _Float16* __restrict__ agg) {
  __shared__ _Float16 Kt[32 * 264];
  __shared__ _Float16 Vt[32 * 264];
  __shared__ float    Sm[16 * 33];
  __shared__ _Float16 Pm[16 * 32];
  __shared__ float qx[16], qy[16], qsq[16];
  __shared__ float rowM[16], rowZ[16], rowScale[16];

  const int tid  = threadIdx.x;
  const int wave = tid >> 5;
  const int l    = tid & 31;
  const int bb   = blockIdx.x >> 7;          // batch
  const int qb   = (blockIdx.x & 127) << 4;  // query base within batch
  const size_t rowbase = (size_t)bb * NPTS + qb;
  const float gab = fabsf(gamma_p[0]);

  if (tid < 16) {
    float x = coords[(rowbase + tid) * 2 + 0];
    float y = coords[(rowbase + tid) * 2 + 1];
    qx[tid] = x; qy[tid] = y; qsq[tid] = x * x + y * y;
    rowM[tid] = -3.0e38f; rowZ[tid] = 0.0f; rowScale[tid] = 0.0f;
  }

  // waves 0,1 cache the whole 16x256 q tile as 8 A-fragments (64 VGPRs)
  v16h qf[8];
  if (wave < 2) {
#pragma unroll
    for (int kd = 0; kd < 8; ++kd)
      qf[kd] = frag_a_f16(q + rowbase * DDIM + kd * 32, DDIM);
  }

  v8f acc0 = {}, acc1 = {};
  const int nb0 = wave * 32;
  const int nb1 = nb0 + 16;
  __syncthreads();

  for (int kb = 0; kb < NPTS; kb += 32) {
    const _Float16* kgbase = kmat + ((size_t)bb * NPTS + kb) * DDIM;
    const _Float16* vgbase = vmat + ((size_t)bb * NPTS + kb) * DDIM;

    // prefetch the NEXT chunk while this one is staged/consumed
    if (kb + 32 < NPTS) {
      __builtin_prefetch((const char*)(kgbase + 32 * DDIM) + (tid << 6), 0, 3);
      __builtin_prefetch((const char*)(vgbase + 32 * DDIM) + (tid << 6), 0, 3);
    }

#if ASYNC_LDS
    // async DMA: 1024 x 16B segments per matrix, 4 per thread per matrix
#pragma unroll
    for (int i = 0; i < 4; ++i) {
      int seg = tid + (i << 8);       // 0..1023
      int r   = seg >> 5;             // key row 0..31
      int s8  = (seg & 31) << 3;      // half offset within row (16B granules)
      __builtin_amdgcn_global_load_async_to_lds_b128(
          (v4i_vs*)(kgbase + (size_t)r * DDIM + s8),
          (lds_v4i_t*)(Kt + r * 264 + s8), 0, 0);
      __builtin_amdgcn_global_load_async_to_lds_b128(
          (v4i_vs*)(vgbase + (size_t)r * DDIM + s8),
          (lds_v4i_t*)(Vt + r * 264 + s8), 0, 0);
    }
    __builtin_amdgcn_s_wait_asynccnt(0);
#else
    // fallback: vectorized B128 copies through VGPRs
    {
      const uint4* kg = (const uint4*)kgbase;
      const uint4* vg = (const uint4*)vgbase;
#pragma unroll
      for (int i = 0; i < 4; ++i) {
        int seg = tid + (i << 8);
        int r = seg >> 5, c = seg & 31;    // 32 x uint4 per row
        *(uint4*)(Kt + r * 264 + c * 8) = kg[r * 32 + c];
        *(uint4*)(Vt + r * 264 + c * 8) = vg[r * 32 + c];
      }
    }
#endif
    __syncthreads();

    // scores: waves 0,1 each produce one 16x16 subtile of S (16x32)
    if (wave < 2) {
      v8f sc = {};
#pragma unroll
      for (int kd = 0; kd < 8; ++kd)
        sc = wmma_f16(qf[kd], frag_b_nk(Kt + (wave * 16) * 264 + kd * 32, 264), sc);
#pragma unroll
      for (int r = 0; r < 8; ++r) sc[r] *= 0.0625f;  // 1/sqrt(256)
      frag_c_store(Sm + wave * 16, 33, sc);
    }
    __syncthreads();

    // online softmax stats per query row
    if (tid < 16) {
      int r = tid;
      float m  = rowM[r];
      float cm = -3.0e38f;
      for (int j = 0; j < 32; ++j) cm = fmaxf(cm, Sm[r * 33 + j]);
      float nm = fmaxf(m, cm);
      float sc = __expf(m - nm);
      float z  = rowZ[r] * sc;
      for (int j = 0; j < 32; ++j) z += __expf(Sm[r * 33 + j] - nm);
      rowZ[r] = z; rowM[r] = nm; rowScale[r] = sc;
    }
    __syncthreads();

    // rescale running accumulators
    {
      const int mbase = (l >> 4) << 3;
#pragma unroll
      for (int r = 0; r < 8; ++r) {
        float s = rowScale[mbase + r];
        acc0[r] *= s; acc1[r] *= s;
      }
    }
    // build P = exp(s-m) * adjacency * keep   (16x32, f16)
    for (int e = tid; e < 512; e += 256) {
      int r = e >> 5, j = e & 31;
      int key = kb + j;
      float xk = coords[((size_t)bb * NPTS + key) * 2 + 0];
      float yk = coords[((size_t)bb * NPTS + key) * 2 + 1];
      float d2 = qsq[r] + xk * xk + yk * yk - 2.0f * (qx[r] * xk + qy[r] * yk);
      float dist = sqrtf(fmaxf(d2, 0.0f));
      float adj  = __expf(-dist * gab);
      float keep = (mask[bb * NPTS + key] == 0) ? 1.0f : 0.0f;
      float p = __expf(Sm[r * 33 + j] - rowM[r]) * adj * keep;
      Pm[r * 32 + j] = (_Float16)p;
    }
    __syncthreads();

    // agg += P @ V_chunk : one 16x16x32 WMMA per owned column tile
    {
      v16h a = frag_a_f16(Pm, 32);
      acc0 = wmma_f16(a, frag_b_kn(Vt + nb0, 264), acc0);
      acc1 = wmma_f16(a, frag_b_kn(Vt + nb1, 264), acc1);
    }
    __syncthreads();
  }

  // finalize: divide by softmax denominator Z and emit f16
  {
    const int n = l & 15;
    const int mbase = (l >> 4) << 3;
#pragma unroll
    for (int r = 0; r < 8; ++r) {
      float iz = 1.0f / rowZ[mbase + r];
      size_t gm = rowbase + mbase + r;
      agg[gm * DDIM + nb0 + n] = (_Float16)(acc0[r] * iz);
      agg[gm * DDIM + nb1 + n] = (_Float16)(acc1[r] * iz);
    }
  }
}

// ---------------------------------------------------------------------------
// K4: out = relu(LN(agg @ bn_w + bn_b)); masked column sums -> pooled_sum
// ---------------------------------------------------------------------------
__global__ __launch_bounds__(256) void k_outproj_pool(
    const _Float16* __restrict__ aggm, const float* __restrict__ bn_w,
    const float* __restrict__ bn_b, const float* __restrict__ bn_g,
    const float* __restrict__ bn_beta, const int* __restrict__ mask,
    float* __restrict__ pooled_sum) {
  __shared__ _Float16 Bs[32 * 136];
  __shared__ float Cs[16 * 136];
  __shared__ float redS[256], redS2[256];
  __shared__ float rowMu[16], rowRs[16];
  __shared__ float keepRow[16];

  const int tid  = threadIdx.x;
  const int wave = tid >> 5;
  const int bb   = blockIdx.x >> 7;
  const int qb   = (blockIdx.x & 127) << 4;
  const size_t rowbase = (size_t)bb * NPTS + qb;

  if (tid < 16) keepRow[tid] = (mask[rowbase + tid] == 0) ? 1.0f : 0.0f;

  v8f acc = {};
  const int nb = wave * 16;
  for (int kk = 0; kk < DDIM; kk += 32) {
    for (int e = tid; e < 32 * BDIM; e += 256) {
      int r = e >> 7, c = e & 127;
      Bs[r * 136 + c] = (_Float16)bn_w[(size_t)(kk + r) * BDIM + c];
    }
    __syncthreads();
    v16h a = frag_a_f16(aggm + rowbase * DDIM + kk, DDIM);
    acc = wmma_f16(a, frag_b_kn(Bs + nb, 136), acc);
    __syncthreads();
  }
  {
    const int n = tid & 15;
    const float bv = bn_b[nb + n];
#pragma unroll
    for (int r = 0; r < 8; ++r) acc[r] += bv;
  }
  frag_c_store(Cs + nb, 136, acc);
  __syncthreads();

  // LayerNorm over 128 per row: 16 threads/row, 8 cols each
  const int r  = tid >> 4;
  const int c0 = (tid & 15) * 8;
  float s = 0.f, s2 = 0.f;
#pragma unroll
  for (int j = 0; j < 8; ++j) {
    float v = Cs[r * 136 + c0 + j];
    s += v; s2 += v * v;
  }
  redS[tid] = s; redS2[tid] = s2;
  __syncthreads();
  if ((tid & 15) == 0) {
    float S = 0.f, S2 = 0.f;
    for (int j = 0; j < 16; ++j) { S += redS[(r << 4) + j]; S2 += redS2[(r << 4) + j]; }
    float mu  = S / (float)BDIM;
    float var = fmaxf(S2 / (float)BDIM - mu * mu, 0.0f);
    rowMu[r] = mu;
    rowRs[r] = rsqrtf(var + 1e-5f);
  }
  __syncthreads();
#pragma unroll
  for (int j = 0; j < 8; ++j) {
    int cc = c0 + j;
    float v = (Cs[r * 136 + cc] - rowMu[r]) * rowRs[r] * bn_g[cc] + bn_beta[cc];
    Cs[r * 136 + cc] = fmaxf(v, 0.0f);
  }
  __syncthreads();

  if (tid < BDIM) {
    float ssum = 0.f;
#pragma unroll
    for (int rr = 0; rr < 16; ++rr) ssum += keepRow[rr] * Cs[rr * 136 + tid];
    atomicAdd(&pooled_sum[bb * BDIM + tid], ssum);
  }
}

// ---------------------------------------------------------------------------
// K5: pooled = pooled_sum / max(count, 1e-9); logits = relu(pooled@c1)+...@c2
// Output layout: d_out[0:80] = logits, d_out[80:1104] = pooled
// ---------------------------------------------------------------------------
__global__ __launch_bounds__(256) void k_head(
    const float* __restrict__ pooled_sum, const int* __restrict__ mask,
    const float* __restrict__ c1_w, const float* __restrict__ c1_b,
    const float* __restrict__ c2_w, const float* __restrict__ c2_b,
    float* __restrict__ out) {
  __shared__ float cbuf[256];
  __shared__ float cnt[BATCH];
  __shared__ float pooled[BATCH * BDIM];
  __shared__ float hid[BATCH * 64];

  const int tid = threadIdx.x;
  const int b  = tid >> 5;
  const int l2 = tid & 31;
  float local = 0.f;
  for (int n = l2; n < NPTS; n += 32)
    local += (mask[b * NPTS + n] == 0) ? 1.0f : 0.0f;
  cbuf[tid] = local;
  __syncthreads();
  if (l2 == 0) {
    float s = 0.f;
    for (int j = 0; j < 32; ++j) s += cbuf[(b << 5) + j];
    cnt[b] = s;
  }
  __syncthreads();

  for (int idx = tid; idx < BATCH * BDIM; idx += 256) {
    int bb = idx >> 7;
    float pv = pooled_sum[idx] / fmaxf(cnt[bb], 1e-9f);
    pooled[idx] = pv;
    out[BATCH * NCLS + idx] = pv;
  }
  __syncthreads();

  for (int idx = tid; idx < BATCH * 64; idx += 256) {
    int bb = idx >> 6, j = idx & 63;
    float s = c1_b[j];
    for (int i = 0; i < BDIM; ++i) s += pooled[bb * BDIM + i] * c1_w[i * 64 + j];
    hid[idx] = fmaxf(s, 0.0f);
  }
  __syncthreads();

  if (tid < BATCH * NCLS) {
    int bb = tid / NCLS, j = tid % NCLS;
    float s = c2_b[j];
    for (int i = 0; i < 64; ++i) s += hid[bb * 64 + i] * c2_w[i * NCLS + j];
    out[tid] = s;
  }
}

// ---------------------------------------------------------------------------
// Launcher
// ---------------------------------------------------------------------------
extern "C" void kernel_launch(void* const* d_in, const int* in_sizes, int n_in,
                              void* d_out, int out_size, void* d_ws, size_t ws_size,
                              hipStream_t stream) {
  (void)in_sizes; (void)n_in; (void)out_size; (void)ws_size;

  const float* lf      = (const float*)d_in[0];
  const float* coords  = (const float*)d_in[1];
  const int*   mask    = (const int*)d_in[2];
  const float* enc_w   = (const float*)d_in[3];
  const float* enc_b   = (const float*)d_in[4];
  const float* enc_g   = (const float*)d_in[5];
  const float* enc_be  = (const float*)d_in[6];
  const float* gamma   = (const float*)d_in[7];
  const float* wq      = (const float*)d_in[8];
  const float* bq      = (const float*)d_in[9];
  const float* wk      = (const float*)d_in[10];
  const float* bk      = (const float*)d_in[11];
  const float* wv      = (const float*)d_in[12];
  const float* bv      = (const float*)d_in[13];
  const float* bn_w    = (const float*)d_in[14];
  const float* bn_b    = (const float*)d_in[15];
  const float* bn_g    = (const float*)d_in[16];
  const float* bn_be   = (const float*)d_in[17];
  const float* c1_w    = (const float*)d_in[18];
  const float* c1_b    = (const float*)d_in[19];
  const float* c2_w    = (const float*)d_in[20];
  const float* c2_b    = (const float*)d_in[21];
  float* outp = (float*)d_out;

  // workspace layout (f16 activations, ~42 MB)
  const size_t ACT = (size_t)MROWS * DDIM * sizeof(_Float16);  // 8 MB each
  char* ws = (char*)d_ws;
  _Float16* h_f16 = (_Float16*)(ws + 0 * ACT);
  _Float16* q_f16 = (_Float16*)(ws + 1 * ACT);
  _Float16* k_f16 = (_Float16*)(ws + 2 * ACT);
  _Float16* v_f16 = (_Float16*)(ws + 3 * ACT);
  _Float16* a_f16 = (_Float16*)(ws + 4 * ACT);
  float*    psum  = (float*)(ws + 5 * ACT);

  const int rowBlocks = MROWS / 16;  // 1024

  k_zero<<<(BATCH * BDIM + 255) / 256, 256, 0, stream>>>(psum, BATCH * BDIM);
  k_encoder<<<rowBlocks, 256, 0, stream>>>(lf, enc_w, enc_b, enc_g, enc_be, h_f16);
  k_linear_qkv<<<rowBlocks, 256, 0, stream>>>(h_f16, wq, bq, q_f16);
  k_linear_qkv<<<rowBlocks, 256, 0, stream>>>(h_f16, wk, bk, k_f16);
  k_linear_qkv<<<rowBlocks, 256, 0, stream>>>(h_f16, wv, bv, v_f16);
  k_attn<<<rowBlocks, 256, 0, stream>>>(q_f16, k_f16, v_f16, coords, mask, gamma, a_f16);
  k_outproj_pool<<<rowBlocks, 256, 0, stream>>>(a_f16, bn_w, bn_b, bn_g, bn_be, mask, psum);
  k_head<<<1, 256, 0, stream>>>(psum, mask, c1_w, c1_b, c2_w, c2_b, outp);
}
